// InstructBertSelfAttention_360777253106
// MI455X (gfx1250) — matrix operand order repaired and checked by reading, not compiled, not run on hardware
//
#include <hip/hip_runtime.h>
#include <hip/hip_bf16.h>
#include <stdint.h>

// ---------------------------------------------------------------------------
// InstructBERT self-attention with relative_key_query position scores.
// MI455X / gfx1250: wave32, WMMA bf16 16x16x32 (f32 accum), flash-style
// attention with a banded dist_emb trick (31-row band per 16x16 tile).
// ---------------------------------------------------------------------------

#define HDIM  16
#define DDIM  64
#define SLEN  1024
#define LLEN  128
#define BATCH 4
#define HID   1024

typedef __attribute__((ext_vector_type(16))) __bf16 bf16x16;
typedef __attribute__((ext_vector_type(8)))  float  f32x8;

union FragB { bf16x16 v; uint32_t u[8]; };

__device__ __forceinline__ uint16_t f2bf(float f) {
  uint32_t u = __float_as_uint(f);
  u += 0x7FFFu + ((u >> 16) & 1u);           // round-to-nearest-even
  return (uint16_t)(u >> 16);
}
__device__ __forceinline__ uint32_t pack2bf(float a, float b) {
  return (uint32_t)f2bf(a) | ((uint32_t)f2bf(b) << 16);
}
__device__ __forceinline__ uint32_t ld32(const uint16_t* p) {
  return *reinterpret_cast<const uint32_t*>(p);
}
__device__ __forceinline__ f32x8 wmma_bf16(const FragB& a, const FragB& b, f32x8 c) {
  return __builtin_amdgcn_wmma_f32_16x16x32_bf16(false, a.v, false, b.v,
                                                 (short)0, c, false, false);
}

// ---------------------------------------------------------------------------
// f32 -> bf16 elementwise (with zero tail padding for dist_emb row 2047)
// ---------------------------------------------------------------------------
__global__ void cvt_kernel(const float* __restrict__ in, uint16_t* __restrict__ out,
                           int n, int n_src) {
  int i = blockIdx.x * 256 + threadIdx.x;
  if (i < n) out[i] = (i < n_src) ? f2bf(in[i]) : (uint16_t)0;
}

// ---------------------------------------------------------------------------
// W [1024 k][1024 n] f32 -> Wt [1024 n][1024 k] bf16 (LDS-tiled transpose)
// ---------------------------------------------------------------------------
__global__ void wt_kernel(const float* __restrict__ W, uint16_t* __restrict__ Wt) {
  __shared__ float tile[32][33];
  int k0 = blockIdx.x * 32, n0 = blockIdx.y * 32;
  int tx = threadIdx.x, ty = threadIdx.y;            // block (32,8)
  for (int i = ty; i < 32; i += 8)
    tile[i][tx] = W[(size_t)(k0 + i) * HID + n0 + tx];
  __syncthreads();
  for (int i = ty; i < 32; i += 8)
    Wt[(size_t)(n0 + i) * HID + k0 + tx] = f2bf(tile[tx][i]);
}

// ---------------------------------------------------------------------------
// Projection GEMM: C = Xb @ Wt^T + bias, written per-head bf16.
// Each wave owns a 16(M) x 128(N) tile; block = 8 waves -> 128(M) x 128(N).
// Per k-step: 1 A-frag (shared by 8 WMMA) + 8 B-frags loaded in one clause,
// then 8 back-to-back WMMAs; next A-frag load overlaps the WMMAs.
// A-frag (16x32 bf16): lane M = lane%16, K = 2v+p + 8*(lane/16) (+8 if v>=4)
// B-frag (32x16 bf16): lane N = lane%16, K = 2v+p + 8*(lane/16) (+8 if v>=4)
// ---------------------------------------------------------------------------
__global__ __launch_bounds__(256) void proj_kernel(
    const uint16_t* __restrict__ Xb,   // [Mrows][1024] bf16
    const uint16_t* __restrict__ Wt,   // [1024 n][1024 k] bf16
    const float*    __restrict__ bias, // [1024]
    uint16_t*       __restrict__ out,  // per-head layout
    int tshift, int transposed)        // Tseq = 1 << tshift (1024 or 128)
{
  const int lane = threadIdx.x & 31, w = threadIdx.x >> 5;
  const int hl = lane >> 4, ln = lane & 15;
  const int m0 = blockIdx.x * 128 + w * 16;
  const int n0 = blockIdx.y * 128;
  const int tmask = (1 << tshift) - 1;

  const uint16_t* arow = Xb + (size_t)(m0 + ln) * HID;
  f32x8 acc[8];
  f32x8 zero8 = {0,0,0,0,0,0,0,0};
#pragma unroll
  for (int c = 0; c < 8; ++c) acc[c] = zero8;

  const int kAoff = 8 * hl;
  FragB a;
#pragma unroll
  for (int v = 0; v < 8; ++v)
    a.u[v] = ld32(arow + 2 * v + kAoff + (v >= 4 ? 8 : 0));

  for (int k0 = 0; k0 < HID; k0 += 32) {
    // load all 8 B-frags first (single long clause), then prefetch next A
    FragB bf[8];
#pragma unroll
    for (int c = 0; c < 8; ++c) {
      const uint16_t* brow = Wt + (size_t)(n0 + c * 16 + ln) * HID + k0;
#pragma unroll
      for (int v = 0; v < 8; ++v) bf[c].u[v] = ld32(brow + 2 * v + kAoff + (v >= 4 ? 8 : 0));
    }
    FragB a_next;
    if (k0 + 32 < HID) {
#pragma unroll
      for (int v = 0; v < 8; ++v)
        a_next.u[v] = ld32(arow + k0 + 32 + 2 * v + kAoff + (v >= 4 ? 8 : 0));
    }
#pragma unroll
    for (int c = 0; c < 8; ++c) acc[c] = wmma_bf16(a, bf[c], acc[c]);
    a = a_next;
  }

  // epilogue: bias add, scatter to per-head layout (bf16); shifts, no division
#pragma unroll
  for (int c = 0; c < 8; ++c) {
    int n = n0 + c * 16 + ln;
    int h = n >> 6, d = n & 63;
    float bv = bias[n];
#pragma unroll
    for (int v = 0; v < 8; ++v) {
      int m = m0 + v + 8 * hl;
      int bb = m >> tshift, t = m & tmask;
      float val = acc[c][v] + bv;
      size_t idx = transposed
          ? (((size_t)((bb * HDIM + h) * DDIM + d) << tshift) + t)   // [B,H,D,T]
          : (((size_t)((bb * HDIM + h) << tshift) + t) * DDIM + d);  // [B,H,T,D]
      out[idx] = f2bf(val);
    }
  }
}

// ---------------------------------------------------------------------------
// Fused attention. Block = 8 waves; wave owns a 16-row l-tile of one (b,h).
// Per 16-col r-tile: 2 WMMA (q.kT) + 8 WMMA (banded positional T/U) +
// 4 WMMA (P.V, K=16 zero-padded to 32), online softmax across 16-lane rows.
// ---------------------------------------------------------------------------
__global__ __launch_bounds__(256) void attn_kernel(
    const uint16_t* __restrict__ q_ws,   // [B,H,S,D] bf16
    const uint16_t* __restrict__ k_ws,   // [B,H,S,D] bf16
    const uint16_t* __restrict__ vT_ws,  // [B,H,D,S] bf16
    const uint16_t* __restrict__ ik_ws,  // [B,H,L,D] bf16
    const uint16_t* __restrict__ ivT_ws, // [B,H,D,L] bf16
    const uint16_t* __restrict__ distb,  // [2048,64] bf16 (row 2047 zero)
    const float*    __restrict__ amask,  // [B,S]
    const float*    __restrict__ imask,  // [B,L]
    const float*    __restrict__ gate,   // [H]
    float*          __restrict__ out)    // [B,S,HID]
{
  const int lane = threadIdx.x & 31, w = threadIdx.x >> 5;
  const int hl = lane >> 4, ln = lane & 15;
  const int b = blockIdx.z, h = blockIdx.y;
  const int l0 = blockIdx.x * 128 + w * 16;
  const float inv_sqrt_d = 0.125f;

  __shared__ float smem[8][1280];        // per wave: T[16*32] U[16*32] P[16*16]
  float* Tl = smem[w];
  float* Ul = Tl + 512;
  float* Pl = Ul + 512;

  // persistent Q A-frags (K = d 0..31 and 32..63)
  const uint16_t* qrow = q_ws + ((size_t)((b * HDIM + h) * SLEN) + l0 + ln) * DDIM;
  FragB qa0, qa1;
#pragma unroll
  for (int v = 0; v < 8; ++v) {
    int kk = 2 * v + 16 * hl;
    qa0.u[v] = ld32(qrow + kk);
    qa1.u[v] = ld32(qrow + 32 + kk);
  }

  f32x8 zero8 = {0,0,0,0,0,0,0,0};

  // ---------------- instruction attention (own softmax, gated) -------------
  f32x8 gctx[4];
  {
    f32x8 ictx[4] = {zero8, zero8, zero8, zero8};
    float mi[8], si[8];
#pragma unroll
    for (int v = 0; v < 8; ++v) { mi[v] = -1e30f; si[v] = 0.f; }

    for (int r0 = 0; r0 < LLEN; r0 += 16) {
      const uint16_t* krow = ik_ws + ((size_t)((b * HDIM + h) * LLEN) + r0 + ln) * DDIM;
      FragB kb0, kb1;
#pragma unroll
      for (int v = 0; v < 8; ++v) {
        int kk = 2 * v + 16 * hl;
        kb0.u[v] = ld32(krow + kk);
        kb1.u[v] = ld32(krow + 32 + kk);
      }
      f32x8 sc = zero8;
      sc = wmma_bf16(qa0, kb0, sc);
      sc = wmma_bf16(qa1, kb1, sc);

      float cmask = imask[b * LLEN + r0 + ln];
      float p[8];
#pragma unroll
      for (int v = 0; v < 8; ++v) {
        float s = sc[v] * inv_sqrt_d + cmask;
        float rm = s;
#pragma unroll
        for (int off = 1; off < 16; off <<= 1)
          rm = fmaxf(rm, __shfl_xor(rm, off, 32));
        float mn  = fmaxf(mi[v], rm);
        float scl = __expf(mi[v] - mn);
        float pe  = __expf(s - mn);
        float rs  = pe;
#pragma unroll
        for (int off = 1; off < 16; off <<= 1)
          rs += __shfl_xor(rs, off, 32);
        si[v] = si[v] * scl + rs;
        mi[v] = mn;
#pragma unroll
        for (int c = 0; c < 4; ++c) ictx[c][v] *= scl;
        p[v] = pe;
      }
#pragma unroll
      for (int v = 0; v < 8; ++v) Pl[(v + 8 * hl) * 16 + ln] = p[v];
      asm volatile("s_wait_dscnt 0" ::: "memory");
      FragB pa;
#pragma unroll
      for (int v = 0; v < 4; ++v) {
        int kk = 2 * v;
        pa.u[v]     = hl == 0 ? pack2bf(Pl[ln * 16 + kk], Pl[ln * 16 + kk + 1]) : 0;
        pa.u[v + 4] = hl == 0 ? pack2bf(Pl[ln * 16 + kk + 8], Pl[ln * 16 + kk + 9]) : 0;  // lanes 16..31: zero-pad K 16..31
      }
#pragma unroll
      for (int c = 0; c < 4; ++c) {
        FragB vb;
        if (hl == 0) {                   // lanes 16..31 carry K=16..31 -> zero
          const uint16_t* vp = ivT_ws +
              ((size_t)((b * HDIM + h) * DDIM) + c * 16 + ln) * LLEN + r0;
#pragma unroll
          for (int v = 0; v < 8; ++v) vb.u[v] = ld32(vp + 2 * v);
        } else {
#pragma unroll
          for (int v = 0; v < 8; ++v) vb.u[v] = 0;
        }
        ictx[c] = wmma_bf16(pa, vb, ictx[c]);
      }
    }
    float g = tanhf(gate[h]);
#pragma unroll
    for (int v = 0; v < 8; ++v) {
      float gi = g / si[v];
#pragma unroll
      for (int c = 0; c < 4; ++c) gctx[c][v] = gi * ictx[c][v];
    }
  }

  // ---------------- main attention with banded relative positions ----------
  f32x8 ctx[4] = {zero8, zero8, zero8, zero8};
  float m2[8], s2[8];
#pragma unroll
  for (int v = 0; v < 8; ++v) { m2[v] = -1e30f; s2[v] = 0.f; }

  for (int r0 = 0; r0 < SLEN; r0 += 16) {
    const uint16_t* krow = k_ws + ((size_t)((b * HDIM + h) * SLEN) + r0 + ln) * DDIM;
    __builtin_prefetch(krow + 16 * DDIM, 0, 1);     // next r-tile -> global_prefetch_b8

    FragB kb0, kb1, ka0, ka1;                       // K as WMMA-B and as WMMA-A
#pragma unroll
    for (int v = 0; v < 8; ++v) {
      int kB = 2 * v + 16 * hl;
      kb0.u[v] = ld32(krow + kB);
      kb1.u[v] = ld32(krow + 32 + kB);
      int kA = 2 * v + 16 * hl;
      ka0.u[v] = ld32(krow + kA);
      ka1.u[v] = ld32(krow + 32 + kA);
    }
    f32x8 sc = zero8;
    sc = wmma_bf16(qa0, kb0, sc);
    sc = wmma_bf16(qa1, kb1, sc);

    // dist_emb band: rows base..base+31 cover all (l-r+1023) of this tile
    const int base = l0 - r0 + 1008;                // in [0, 2016]
    const uint16_t* dr0 = distb + (size_t)(base + ln) * DDIM;
    const uint16_t* dr1 = distb + (size_t)(base + 16 + ln) * DDIM;
    FragB d00, d01, d10, d11;
#pragma unroll
    for (int v = 0; v < 8; ++v) {
      int kB = 2 * v + 16 * hl;
      d00.u[v] = ld32(dr0 + kB);  d01.u[v] = ld32(dr0 + 32 + kB);
      d10.u[v] = ld32(dr1 + kB);  d11.u[v] = ld32(dr1 + 32 + kB);
    }
    f32x8 T0 = zero8, T1 = zero8, U0 = zero8, U1 = zero8;
    T0 = wmma_bf16(qa0, d00, T0); T0 = wmma_bf16(qa1, d01, T0);
    T1 = wmma_bf16(qa0, d10, T1); T1 = wmma_bf16(qa1, d11, T1);
    U0 = wmma_bf16(ka0, d00, U0); U0 = wmma_bf16(ka1, d01, U0);
    U1 = wmma_bf16(ka0, d10, U1); U1 = wmma_bf16(ka1, d11, U1);

#pragma unroll
    for (int v = 0; v < 8; ++v) {
      int cr = v + 8 * hl;
      Tl[cr * 32 + ln]      = T0[v];
      Tl[cr * 32 + 16 + ln] = T1[v];
      Ul[cr * 32 + ln]      = U0[v];
      Ul[cr * 32 + 16 + ln] = U1[v];
    }
    asm volatile("s_wait_dscnt 0" ::: "memory");

    float cmask = amask[b * SLEN + r0 + ln];
    float p[8];
#pragma unroll
    for (int v = 0; v < 8; ++v) {
      int li = v + 8 * hl;
      int j  = li - ln + 15;                        // diagonal gather, 0..30
      float s = (sc[v] + Tl[li * 32 + j] + Ul[ln * 32 + j]) * inv_sqrt_d + cmask;
      float rm = s;
#pragma unroll
      for (int off = 1; off < 16; off <<= 1)
        rm = fmaxf(rm, __shfl_xor(rm, off, 32));
      float mn  = fmaxf(m2[v], rm);
      float scl = __expf(m2[v] - mn);
      float pe  = __expf(s - mn);
      float rs  = pe;
#pragma unroll
      for (int off = 1; off < 16; off <<= 1)
        rs += __shfl_xor(rs, off, 32);
      s2[v] = s2[v] * scl + rs;
      m2[v] = mn;
#pragma unroll
      for (int c = 0; c < 4; ++c) ctx[c][v] *= scl;
      p[v] = pe;
    }
#pragma unroll
    for (int v = 0; v < 8; ++v) Pl[(v + 8 * hl) * 16 + ln] = p[v];
    asm volatile("s_wait_dscnt 0" ::: "memory");
    FragB pa;
#pragma unroll
    for (int v = 0; v < 4; ++v) {
      int kk = 2 * v;
      pa.u[v]     = hl == 0 ? pack2bf(Pl[ln * 16 + kk], Pl[ln * 16 + kk + 1]) : 0;
      pa.u[v + 4] = hl == 0 ? pack2bf(Pl[ln * 16 + kk + 8], Pl[ln * 16 + kk + 9]) : 0;
    }
#pragma unroll
    for (int c = 0; c < 4; ++c) {
      FragB vb;
      if (hl == 0) {
        const uint16_t* vp = vT_ws +
            ((size_t)((b * HDIM + h) * DDIM) + c * 16 + ln) * SLEN + r0;
#pragma unroll
        for (int v = 0; v < 8; ++v) vb.u[v] = ld32(vp + 2 * v);
      } else {
#pragma unroll
        for (int v = 0; v < 8; ++v) vb.u[v] = 0;
      }
      ctx[c] = wmma_bf16(pa, vb, ctx[c]);
    }
  }

  // epilogue: normalize, add gated instruction context, write [B,S,HID] f32
#pragma unroll
  for (int v = 0; v < 8; ++v) {
    float inv = 1.0f / s2[v];
    int li = v + 8 * hl;
#pragma unroll
    for (int c = 0; c < 4; ++c) {
      int d = c * 16 + ln;
      float val = ctx[c][v] * inv + gctx[c][v];
      out[(size_t)(b * SLEN + l0 + li) * HID + h * DDIM + d] = val;
    }
  }
}

// ---------------------------------------------------------------------------
extern "C" void kernel_launch(void* const* d_in, const int* in_sizes, int n_in,
                              void* d_out, int out_size, void* d_ws, size_t ws_size,
                              hipStream_t stream) {
  (void)in_sizes; (void)n_in; (void)out_size; (void)ws_size;
  const float* hs    = (const float*)d_in[0];
  const float* amask = (const float*)d_in[1];
  const float* ihs   = (const float*)d_in[2];
  const float* imask = (const float*)d_in[3];
  const float* Wq    = (const float*)d_in[4];
  const float* bq    = (const float*)d_in[5];
  const float* Wk    = (const float*)d_in[6];
  const float* bk    = (const float*)d_in[7];
  const float* Wv    = (const float*)d_in[8];
  const float* bv    = (const float*)d_in[9];
  const float* gate  = (const float*)d_in[10];
  const float* dist  = (const float*)d_in[11];
  float* out = (float*)d_out;

  char* ws = (char*)d_ws;
  size_t off = 0;
  auto alloc = [&](size_t bytes) -> char* {
    char* p = ws + off;
    off += (bytes + 255) & ~(size_t)255;
    return p;
  };
  uint16_t* xb_h   = (uint16_t*)alloc((size_t)4096 * 1024 * 2);
  uint16_t* xb_i   = (uint16_t*)alloc((size_t)512 * 1024 * 2);
  uint16_t* Wqt    = (uint16_t*)alloc((size_t)1024 * 1024 * 2);
  uint16_t* Wkt    = (uint16_t*)alloc((size_t)1024 * 1024 * 2);
  uint16_t* Wvt    = (uint16_t*)alloc((size_t)1024 * 1024 * 2);
  uint16_t* distb  = (uint16_t*)alloc((size_t)2048 * 64 * 2);
  uint16_t* q_ws   = (uint16_t*)alloc((size_t)BATCH * HDIM * SLEN * DDIM * 2);
  uint16_t* k_ws   = (uint16_t*)alloc((size_t)BATCH * HDIM * SLEN * DDIM * 2);
  uint16_t* vT_ws  = (uint16_t*)alloc((size_t)BATCH * HDIM * SLEN * DDIM * 2);
  uint16_t* ik_ws  = (uint16_t*)alloc((size_t)BATCH * HDIM * LLEN * DDIM * 2);
  uint16_t* ivT_ws = (uint16_t*)alloc((size_t)BATCH * HDIM * LLEN * DDIM * 2);

  // 1) bf16 conversions
  {
    int n = 4096 * 1024;
    cvt_kernel<<<(n + 255) / 256, 256, 0, stream>>>(hs, xb_h, n, n);
  }
  {
    int n = 512 * 1024;
    cvt_kernel<<<(n + 255) / 256, 256, 0, stream>>>(ihs, xb_i, n, n);
  }
  {
    int n = 2048 * 64, ns = 2047 * 64;
    cvt_kernel<<<(n + 255) / 256, 256, 0, stream>>>(dist, distb, n, ns);
  }

  // 2) weight transpose+convert: [K,N] f32 -> [N,K] bf16
  dim3 wtg(32, 32), wtb(32, 8);
  wt_kernel<<<wtg, wtb, 0, stream>>>(Wq, Wqt);
  wt_kernel<<<wtg, wtb, 0, stream>>>(Wk, Wkt);
  wt_kernel<<<wtg, wtb, 0, stream>>>(Wv, Wvt);

  // 3) projections (bf16 WMMA GEMM); tshift = log2(Tseq)
  dim3 pg_h(32, 8), pg_i(4, 8);
  proj_kernel<<<pg_h, 256, 0, stream>>>(xb_h, Wqt, bq, q_ws, 10, 0);
  proj_kernel<<<pg_h, 256, 0, stream>>>(xb_h, Wkt, bk, k_ws, 10, 0);
  proj_kernel<<<pg_h, 256, 0, stream>>>(xb_h, Wvt, bv, vT_ws, 10, 1);
  proj_kernel<<<pg_i, 256, 0, stream>>>(xb_i, Wkt, bk, ik_ws, 7, 0);
  proj_kernel<<<pg_i, 256, 0, stream>>>(xb_i, Wvt, bv, ivT_ws, 7, 1);

  // 4) fused flash attention with banded relative positions + gated instr attn
  dim3 ag(SLEN / 128, HDIM, BATCH);
  attn_kernel<<<ag, 256, 0, stream>>>(q_ws, k_ws, vT_ws, ik_ws, ivT_ws, distb,
                                      amask, imask, gate, out);
}
